// RosaTransformer_34359738368452
// MI455X (gfx1250) — compile-verified, matrix operand order checked by reading, not run on hardware
//
#include <hip/hip_runtime.h>

// ======================== CDNA5 / gfx1250 types ============================
typedef __attribute__((ext_vector_type(16))) __bf16 v16bf;
typedef __attribute__((ext_vector_type(8)))  __bf16 v8bf;
typedef __attribute__((ext_vector_type(8)))  float  v8f;
typedef __attribute__((ext_vector_type(4)))  float  f4;
typedef __attribute__((ext_vector_type(4)))  unsigned short us4;

namespace cfg {
constexpr int B   = 8;
constexpr int SEQ = 2048;   // encoder sequence
constexpr int SDQ = 1024;   // decoder sequence
constexpr int IND = 1280;
constexpr int D   = 512;
constexpr int DEPTH = 4;
constexpr int H   = 8;
constexpr int DHD = 64;
constexpr int MF  = 266;    // FAVOR features (logical)
constexpr int MFP = 320;    // FAVOR features padded to a multiple of 64 (zero pad)
constexpr int FFD = 2048;
constexpr int NB  = 128;
}

// ======================== small device helpers =============================
__device__ __forceinline__ unsigned short f2bf(float f) {
  unsigned u = __float_as_uint(f);
  u += 0x7fffu + ((u >> 16) & 1u);      // round-to-nearest-even
  return (unsigned short)(u >> 16);
}
__device__ __forceinline__ float gelu_f(float x) {
  return 0.5f * x * (1.0f + erff(x * 0.70710678118654752440f));
}
__device__ __forceinline__ float blk_sum(float v, float* red) {
  int tid = threadIdx.x;
  red[tid] = v; __syncthreads();
  for (int s = 128; s > 0; s >>= 1) { if (tid < s) red[tid] += red[tid + s]; __syncthreads(); }
  float r = red[0]; __syncthreads();
  return r;
}
__device__ __forceinline__ float blk_max(float v, float* red) {
  int tid = threadIdx.x;
  red[tid] = v; __syncthreads();
  for (int s = 128; s > 0; s >>= 1) { if (tid < s) red[tid] = fmaxf(red[tid], red[tid + s]); __syncthreads(); }
  float r = red[0]; __syncthreads();
  return r;
}
// order-preserving float<->uint for global atomic max
__device__ __forceinline__ unsigned fenc(float f) {
  unsigned u = __float_as_uint(f);
  return (u & 0x80000000u) ? ~u : (u | 0x80000000u);
}
__device__ __forceinline__ float fdec(unsigned e) {
  return (e & 0x80000000u) ? __uint_as_float(e & 0x7fffffffu) : __uint_as_float(~e);
}

// ======================== WMMA GEMM ========================================
// C[M,N] = act(alpha*opA(A) @ opB(B) + bias) * rowScale + beta*C
// opA(A): TA=0 -> A[row*lda+k]   TA=1 -> A[k*lda+row]
// opB(B): TB=0 -> B[k*ldb+n]     TB=1 -> B[n*ldb+k]
// REQUIREMENTS (guaranteed by the launcher):
//   K % 64 == 0, lda % 4 == 0, ldb % 4 == 0, all base pointers 16B aligned.
// M/N edges handled with clamped (branch-free) loads + guarded epilogue.
struct GemmParams {
  const float* A; const float* Bm; const float* bias; const float* rowScale; float* C;
  int M, N, K, G2;
  long lda, ldb, ldc;
  long sA1, sA2, sB1, sB2, sC1, sC2, sRS1, sRS2;
  float alpha, beta;
};

#define GT_BM 128
#define GT_BN 64
#define GT_BK 64
#define GT_LS 72   // LDS row stride in halfs (64 + 8 pad, rows stay 16B aligned)

// One 16x32 bf16 fragment (A-layout per CDNA5 ISA 7.12.2):
// lane l<16 : row M=l,    halves 0..7 -> K=0..7,  halves 8..15 -> K=16..23
// lane l>=16: row M=l-16, halves 0..7 -> K=8..15, halves 8..15 -> K=24..31
__device__ __forceinline__ v16bf load_frag(const unsigned short* base, int lane) {
  int r  = lane & 15;
  int hi = lane >> 4;
  const unsigned short* p = base + r * GT_LS;
  v8bf lo = *(const v8bf*)(p + hi * 8);
  v8bf hv = *(const v8bf*)(p + 16 + hi * 8);
  v16bf out;
#pragma unroll
  for (int i = 0; i < 8; ++i) { out[i] = lo[i]; out[i + 8] = hv[i]; }
  return out;
}

template <int TA, int TB, int ACT>
__global__ __launch_bounds__(256) void gemm_wmma(GemmParams p) {
  __shared__ unsigned short As[2][GT_BM * GT_LS];  // [m][k]
  __shared__ unsigned short Bs[2][GT_BN * GT_LS];  // [n][k]

  const int tid  = threadIdx.x;
  const int lane = tid & 31;
  const int wave = tid >> 5;        // 8 wave32 waves
  const int wm   = wave >> 1;       // 0..3  (M direction)
  const int wn   = wave & 1;        // 0..1  (N direction)

  const int z  = blockIdx.z;
  const int g1 = z / p.G2, g2 = z % p.G2;
  const float* Ag = p.A  + (long)g1 * p.sA1 + (long)g2 * p.sA2;
  const float* Bg = p.Bm + (long)g1 * p.sB1 + (long)g2 * p.sB2;
  float*       Cg = p.C  + (long)g1 * p.sC1 + (long)g2 * p.sC2;
  const float* rs = p.rowScale ? (p.rowScale + (long)g1 * p.sRS1 + (long)g2 * p.sRS2) : nullptr;

  const int m0 = blockIdx.y * GT_BM;
  const int n0 = blockIdx.x * GT_BN;
  const bool fullA = (m0 + GT_BM <= p.M);   // K is always complete
  const bool fullB = (n0 + GT_BN <= p.N);

  float ra[32];   // staged A chunk: 128x64 f32 / 256 threads
  float rb[16];   // staged B chunk: 64x64  f32 / 256 threads

  // ---- global -> registers -----------------------------------------------
  auto loadA = [&](int k0) {
    if (TA == 0) {                         // A[row][k]
      if (fullA) {
#pragma unroll
        for (int it = 0; it < 8; ++it) {
          int q = tid + it * 256;          // 0..2047
          int row = q >> 4;                // 0..127
          int kc  = (q & 15) << 2;         // 0..60
          const float* src = Ag + (long)(m0 + row) * p.lda + (k0 + kc);
          if (it == 0) __builtin_prefetch(src + 2 * GT_BK, 0, 0);  // global_prefetch_b8
          f4 v = *(const f4*)src;
#pragma unroll
          for (int j = 0; j < 4; ++j) ra[it * 4 + j] = v[j];
        }
      } else {                              // clamped, branch-free
#pragma unroll
        for (int it = 0; it < 8; ++it) {
          int q = tid + it * 256;
          int row = q >> 4;
          int kc  = (q & 15) << 2;
          bool ok = (m0 + row) < p.M;
          int rr  = ok ? (m0 + row) : (p.M - 1);
          const float* src = Ag + (long)rr * p.lda + (k0 + kc);
#pragma unroll
          for (int j = 0; j < 4; ++j) ra[it * 4 + j] = ok ? src[j] : 0.f;
        }
      }
    } else {                                // A[k][row]  (64 k-rows x 128 m)
      if (fullA) {
#pragma unroll
        for (int it = 0; it < 8; ++it) {
          int q = tid + it * 256;
          int krow = q >> 5;                // 0..63
          int mm   = (q & 31) << 2;         // 0..124
          f4 v = *(const f4*)(Ag + (long)(k0 + krow) * p.lda + (m0 + mm));
#pragma unroll
          for (int j = 0; j < 4; ++j) ra[it * 4 + j] = v[j];
        }
      } else {
#pragma unroll
        for (int it = 0; it < 8; ++it) {
          int q = tid + it * 256;
          int krow = q >> 5;
          int mm   = (q & 31) << 2;
          const float* src = Ag + (long)(k0 + krow) * p.lda;
#pragma unroll
          for (int j = 0; j < 4; ++j) {
            bool ok = (m0 + mm + j) < p.M;
            int cc  = ok ? (m0 + mm + j) : (p.M - 1);
            ra[it * 4 + j] = ok ? src[cc] : 0.f;
          }
        }
      }
    }
  };
  auto loadB = [&](int k0) {
    if (TB == 0) {                          // B[k][n]  (64 k-rows x 64 n)
      if (fullB) {
#pragma unroll
        for (int it = 0; it < 4; ++it) {
          int q = tid + it * 256;           // 0..1023
          int krow = q >> 4;                // 0..63
          int nn   = (q & 15) << 2;         // 0..60
          f4 v = *(const f4*)(Bg + (long)(k0 + krow) * p.ldb + (n0 + nn));
#pragma unroll
          for (int j = 0; j < 4; ++j) rb[it * 4 + j] = v[j];
        }
      } else {
#pragma unroll
        for (int it = 0; it < 4; ++it) {
          int q = tid + it * 256;
          int krow = q >> 4;
          int nn   = (q & 15) << 2;
          const float* src = Bg + (long)(k0 + krow) * p.ldb;
#pragma unroll
          for (int j = 0; j < 4; ++j) {
            bool ok = (n0 + nn + j) < p.N;
            int cc  = ok ? (n0 + nn + j) : (p.N - 1);
            rb[it * 4 + j] = ok ? src[cc] : 0.f;
          }
        }
      }
    } else {                                // B[n][k]  (64 n-rows x 64 k)
      if (fullB) {
#pragma unroll
        for (int it = 0; it < 4; ++it) {
          int q = tid + it * 256;
          int nrow = q >> 4;                // 0..63
          int kc   = (q & 15) << 2;
          f4 v = *(const f4*)(Bg + (long)(n0 + nrow) * p.ldb + (k0 + kc));
#pragma unroll
          for (int j = 0; j < 4; ++j) rb[it * 4 + j] = v[j];
        }
      } else {
#pragma unroll
        for (int it = 0; it < 4; ++it) {
          int q = tid + it * 256;
          int nrow = q >> 4;
          int kc   = (q & 15) << 2;
          bool ok = (n0 + nrow) < p.N;
          int rr  = ok ? (n0 + nrow) : (p.N - 1);
          const float* src = Bg + (long)rr * p.ldb + (k0 + kc);
#pragma unroll
          for (int j = 0; j < 4; ++j) rb[it * 4 + j] = ok ? src[j] : 0.f;
        }
      }
    }
  };
  // ---- registers -> LDS (fp32 -> bf16) -----------------------------------
  auto storeA = [&](int buf) {
    if (TA == 0) {
#pragma unroll
      for (int it = 0; it < 8; ++it) {
        int q = tid + it * 256;
        int row = q >> 4;
        int kc  = (q & 15) << 2;
        us4 h;
#pragma unroll
        for (int j = 0; j < 4; ++j) h[j] = f2bf(ra[it * 4 + j] * p.alpha);
        *(us4*)&As[buf][row * GT_LS + kc] = h;
      }
    } else {
#pragma unroll
      for (int it = 0; it < 8; ++it) {
        int q = tid + it * 256;
        int krow = q >> 5;
        int mm   = (q & 31) << 2;
#pragma unroll
        for (int j = 0; j < 4; ++j)
          As[buf][(mm + j) * GT_LS + krow] = f2bf(ra[it * 4 + j] * p.alpha);
      }
    }
  };
  auto storeB = [&](int buf) {
    if (TB == 0) {                          // transpose into [n][k]
#pragma unroll
      for (int it = 0; it < 4; ++it) {
        int q = tid + it * 256;
        int krow = q >> 4;
        int nn   = (q & 15) << 2;
#pragma unroll
        for (int j = 0; j < 4; ++j)
          Bs[buf][(nn + j) * GT_LS + krow] = f2bf(rb[it * 4 + j]);
      }
    } else {                                // already k-contiguous
#pragma unroll
      for (int it = 0; it < 4; ++it) {
        int q = tid + it * 256;
        int nrow = q >> 4;
        int kc   = (q & 15) << 2;
        us4 h;
#pragma unroll
        for (int j = 0; j < 4; ++j) h[j] = f2bf(rb[it * 4 + j]);
        *(us4*)&Bs[buf][nrow * GT_LS + kc] = h;
      }
    }
  };

  v8f zero = {0.f, 0.f, 0.f, 0.f, 0.f, 0.f, 0.f, 0.f};
  v8f acc[2][2] = {{zero, zero}, {zero, zero}};

  // ---- software-pipelined main loop (double-buffered LDS) ----------------
  loadA(0); loadB(0); storeA(0); storeB(0);
  __syncthreads();
  const int nk = p.K >> 6;                  // K chunks of 64 (K % 64 == 0)
  for (int c = 0; c < nk; ++c) {
    const int cur = c & 1;
    if (c + 1 < nk) { loadA((c + 1) << 6); loadB((c + 1) << 6); }
    // 2 x (2x2) WMMA from the current buffer
#pragma unroll
    for (int kk = 0; kk < 2; ++kk) {
      const unsigned short* Ab = &As[cur][0] + kk * 32;
      const unsigned short* Bb = &Bs[cur][0] + kk * 32;
      v16bf a0 = load_frag(Ab + (wm * 32 +  0) * GT_LS, lane);
      v16bf a1 = load_frag(Ab + (wm * 32 + 16) * GT_LS, lane);
      v16bf b0 = load_frag(Bb + (wn * 32 +  0) * GT_LS, lane);
      v16bf b1 = load_frag(Bb + (wn * 32 + 16) * GT_LS, lane);
      acc[0][0] = __builtin_amdgcn_wmma_f32_16x16x32_bf16(false, a0, false, b0, (short)0, acc[0][0], false, false);
      acc[0][1] = __builtin_amdgcn_wmma_f32_16x16x32_bf16(false, a0, false, b1, (short)0, acc[0][1], false, false);
      acc[1][0] = __builtin_amdgcn_wmma_f32_16x16x32_bf16(false, a1, false, b0, (short)0, acc[1][0], false, false);
      acc[1][1] = __builtin_amdgcn_wmma_f32_16x16x32_bf16(false, a1, false, b1, (short)0, acc[1][1], false, false);
    }
    if (c + 1 < nk) { storeA(cur ^ 1); storeB(cur ^ 1); }
    __syncthreads();
  }

  // ---- epilogue (C/D layout: N = lane%16, M = r + 8*(lane/16)) -----------
  const int cn = lane & 15;
  const int cr = lane >> 4;
  if (fullA && fullB) {
#pragma unroll
    for (int mt = 0; mt < 2; ++mt)
#pragma unroll
      for (int nt = 0; nt < 2; ++nt)
#pragma unroll
        for (int r = 0; r < 8; ++r) {
          int row = m0 + wm * 32 + mt * 16 + r + cr * 8;
          int col = n0 + wn * 32 + nt * 16 + cn;
          float v = acc[mt][nt][r];
          if (p.bias) v += p.bias[col];
          if (ACT == 1) v = gelu_f(v);
          if (rs) v *= rs[row];
          float* cp = Cg + (long)row * p.ldc + col;
          if (p.beta != 0.f) v += p.beta * (*cp);
          *cp = v;
        }
  } else {
#pragma unroll
    for (int mt = 0; mt < 2; ++mt)
#pragma unroll
      for (int nt = 0; nt < 2; ++nt)
#pragma unroll
        for (int r = 0; r < 8; ++r) {
          int row = m0 + wm * 32 + mt * 16 + r + cr * 8;
          int col = n0 + wn * 32 + nt * 16 + cn;
          if (row < p.M && col < p.N) {
            float v = acc[mt][nt][r];
            if (p.bias) v += p.bias[col];
            if (ACT == 1) v = gelu_f(v);
            if (rs) v *= rs[row];
            float* cp = Cg + (long)row * p.ldc + col;
            if (p.beta != 0.f) v += p.beta * (*cp);
            *cp = v;
          }
        }
  }
}

// ======================== elementwise kernels ==============================
// x = LN(x + sinusoidal(expr)) in place; row = b*SEQ+s, D=512, 256 threads.
__global__ __launch_bounds__(256) void sinus_add_ln_kernel(float* x, const int* expr,
                                                           const float* g, const float* b) {
  __shared__ float red[256];
  long row = blockIdx.x;
  int tid  = threadIdx.x;
  float e  = (float)expr[row];
  float fr = expf(-logf(10000.f) * (float)tid / 256.f);
  float ang = e * fr;
  float t0 = x[row * 512 + tid]       + sinf(ang);
  float t1 = x[row * 512 + tid + 256] + cosf(ang);
  float mu = blk_sum(t0 + t1, red) * (1.f / 512.f);
  float d0 = t0 - mu, d1 = t1 - mu;
  float var = blk_sum(d0 * d0 + d1 * d1, red) * (1.f / 512.f);
  float inv = rsqrtf(var + 1e-5f);
  x[row * 512 + tid]       = d0 * inv * g[tid]       + b[tid];
  x[row * 512 + tid + 256] = d1 * inv * g[tid + 256] + b[tid + 256];
}

// y = LN(x [+ res]); D=512 fixed, one block per row.
__global__ __launch_bounds__(256) void layernorm_kernel(const float* x, const float* res,
                                                        const float* g, const float* b, float* y) {
  __shared__ float red[256];
  long row = blockIdx.x;
  int tid  = threadIdx.x;
  float t0 = x[row * 512 + tid];
  float t1 = x[row * 512 + tid + 256];
  if (res) { t0 += res[row * 512 + tid]; t1 += res[row * 512 + tid + 256]; }
  float mu = blk_sum(t0 + t1, red) * (1.f / 512.f);
  float d0 = t0 - mu, d1 = t1 - mu;
  float var = blk_sum(d0 * d0 + d1 * d1, red) * (1.f / 512.f);
  float inv = rsqrtf(var + 1e-5f);
  y[row * 512 + tid]       = d0 * inv * g[tid]       + b[tid];
  y[row * 512 + tid + 256] = d1 * inv * g[tid + 256] + b[tid + 256];
}

// qp row transform: qp = ratio*(exp(qd - q_diag - rowmax(qd)) + eps); zero pad cols.
__global__ __launch_bounds__(256) void favor_q_kernel(const float* q, float* qp) {
  __shared__ float red[256];
  long idx = blockIdx.x;                  // (b*H+h)*SEQ + s
  int  s   = (int)(idx % cfg::SEQ);
  long bh  = idx / cfg::SEQ;
  int  h   = (int)(bh % cfg::H);
  long b   = bh / cfg::H;
  int tid  = threadIdx.x;
  const float* qrow = q + (b * cfg::SEQ + s) * (long)cfg::D + h * cfg::DHD;
  float qq = 0.f;
  if (tid < cfg::DHD) { float v = qrow[tid]; qq = v * v; }
  float diag = 0.5f * blk_sum(qq, red) * 0.125f;          // dn*dn = DH^-0.5
  float* r = qp + idx * (long)cfg::MFP;
  float v0 = (tid       < cfg::MF) ? r[tid]       : -1e30f;
  float v1 = (tid + 256 < cfg::MF) ? r[tid + 256] : -1e30f;
  float rmax  = blk_max(fmaxf(v0, v1), red);
  float ratio = rsqrtf((float)cfg::MF);
  if (tid < cfg::MF) r[tid] = ratio * (expf(v0 - diag - rmax) + 1e-4f);
  if (tid + 256 < cfg::MF)        r[tid + 256] = ratio * (expf(v1 - diag - rmax) + 1e-4f);
  else if (tid + 256 < cfg::MFP)  r[tid + 256] = 0.f;     // zero K padding
}

__global__ void gmax_init(unsigned* gm) { *gm = 0u; }

// global max over the *logical* (unpadded) kd entries
__global__ __launch_bounds__(256) void kmax_kernel(const float* kd, int n, unsigned* gm) {
  __shared__ float red[256];
  int stride = gridDim.x * 256;
  float m = -1e30f;
  for (int j = blockIdx.x * 256 + threadIdx.x; j < n; j += stride) {
    int row = j / cfg::MF, col = j % cfg::MF;
    m = fmaxf(m, kd[(long)row * cfg::MFP + col]);
  }
  m = blk_max(m, red);
  if (threadIdx.x == 0) atomicMax(gm, fenc(m));
}

// kp = ratio*(exp(kd - k_diag - globalmax) + eps); zero pad cols.
__global__ __launch_bounds__(256) void favor_k_kernel(const float* k, float* kp, const unsigned* gm) {
  __shared__ float red[256];
  long idx = blockIdx.x;
  int  s   = (int)(idx % cfg::SEQ);
  long bh  = idx / cfg::SEQ;
  int  h   = (int)(bh % cfg::H);
  long b   = bh / cfg::H;
  int tid  = threadIdx.x;
  const float* krow = k + (b * cfg::SEQ + s) * (long)cfg::D + h * cfg::DHD;
  float kk = 0.f;
  if (tid < cfg::DHD) { float v = krow[tid]; kk = v * v; }
  float diag = 0.5f * blk_sum(kk, red) * 0.125f;
  float gmax = fdec(*gm);
  float* r = kp + idx * (long)cfg::MFP;
  float ratio = rsqrtf((float)cfg::MF);
  if (tid < cfg::MF) r[tid] = ratio * (expf(r[tid] - diag - gmax) + 1e-4f);
  if (tid + 256 < cfg::MF)        r[tid + 256] = ratio * (expf(r[tid + 256] - diag - gmax) + 1e-4f);
  else if (tid + 256 < cfg::MFP)  r[tid + 256] = 0.f;
}

// k_sum[bh,m] = sum_s kp[bh,s,m]; one block per bh, thread = m.
__global__ __launch_bounds__(512) void ksum_kernel(const float* kp, float* ks) {
  long bh = blockIdx.x;
  int  m  = threadIdx.x;
  if (m >= cfg::MF) return;
  const float* base = kp + bh * (long)cfg::SEQ * cfg::MFP + m;
  float s = 0.f;
  for (int i = 0; i < cfg::SEQ; ++i) s += base[(long)i * cfg::MFP];
  ks[bh * cfg::MF + m] = s;
}

// d_inv[idx] = 1 / dot(qp_row, k_sum)
__global__ __launch_bounds__(256) void dinv_kernel(const float* qp, const float* ks, float* di) {
  __shared__ float red[256];
  long idx = blockIdx.x;
  long bh  = idx / cfg::SEQ;
  const float* r = qp + idx * (long)cfg::MFP;
  const float* K = ks + bh * cfg::MF;
  int tid = threadIdx.x;
  float s = 0.f;
  if (tid       < cfg::MF) s += r[tid]       * K[tid];
  if (tid + 256 < cfg::MF) s += r[tid + 256] * K[tid + 256];
  s = blk_sum(s, red);
  if (tid == 0) di[idx] = 1.f / s;
}

// row softmax over SEQ=2048, in place; 8 elems per thread.
__global__ __launch_bounds__(256) void softmax_kernel(float* x) {
  __shared__ float red[256];
  long row = blockIdx.x;
  float* r = x + row * (long)cfg::SEQ;
  int tid = threadIdx.x;
  float v[8];
  float m = -1e30f;
#pragma unroll
  for (int i = 0; i < 8; ++i) { v[i] = r[tid + i * 256]; m = fmaxf(m, v[i]); }
  m = blk_max(m, red);
  float s = 0.f;
#pragma unroll
  for (int i = 0; i < 8; ++i) { v[i] = expf(v[i] - m); s += v[i]; }
  s = blk_sum(s, red);
  float inv = 1.f / s;
#pragma unroll
  for (int i = 0; i < 8; ++i) r[tid + i * 256] = v[i] * inv;
}

// ======================== host-side launch helpers =========================
static GemmParams gp(const float* A, const float* Bm, const float* bias, const float* rs, float* C,
                     int M, int N, int K, long lda, long ldb, long ldc,
                     float alpha = 1.f, float beta = 0.f, int G2 = 1,
                     long sA1 = 0, long sA2 = 0, long sB1 = 0, long sB2 = 0,
                     long sC1 = 0, long sC2 = 0, long sRS1 = 0, long sRS2 = 0) {
  GemmParams p{A, Bm, bias, rs, C, M, N, K, G2, lda, ldb, ldc,
               sA1, sA2, sB1, sB2, sC1, sC2, sRS1, sRS2, alpha, beta};
  return p;
}

static void launch_gemm(hipStream_t st, int TA, int TB, int ACT, const GemmParams& p, int G1) {
  dim3 grid((p.N + GT_BN - 1) / GT_BN, (p.M + GT_BM - 1) / GT_BM, G1 * p.G2);
  dim3 blk(256, 1, 1);
  if (TA == 0 && TB == 0 && ACT == 0)      gemm_wmma<0, 0, 0><<<grid, blk, 0, st>>>(p);
  else if (TA == 0 && TB == 0 && ACT == 1) gemm_wmma<0, 0, 1><<<grid, blk, 0, st>>>(p);
  else if (TA == 0 && TB == 1)             gemm_wmma<0, 1, 0><<<grid, blk, 0, st>>>(p);
  else                                     gemm_wmma<1, 0, 0><<<grid, blk, 0, st>>>(p);
}

extern "C" void kernel_launch(void* const* d_in, const int* in_sizes, int n_in,
                              void* d_out, int out_size, void* d_ws, size_t ws_size,
                              hipStream_t stream) {
  using namespace cfg;
  (void)in_sizes; (void)n_in; (void)out_size; (void)ws_size;

  // -------- inputs (setup_inputs dict order) --------
  const float* in_enc    = (const float*)d_in[0];
  const float* in_dec    = (const float*)d_in[1];
  const int*   expr      = (const int*)  d_in[2];
  const float* var_W     = (const float*)d_in[3];
  const float* var_b     = (const float*)d_in[4];
  const float* enc_g     = (const float*)d_in[5];
  const float* enc_b     = (const float*)d_in[6];
  const float* dec_g     = (const float*)d_in[7];
  const float* dec_b     = (const float*)d_in[8];
  const float* p_ln1_g   = (const float*)d_in[9];
  const float* p_ln1_b   = (const float*)d_in[10];
  const float* p_Wq      = (const float*)d_in[11];
  const float* p_Wk      = (const float*)d_in[12];
  const float* p_Wv      = (const float*)d_in[13];
  const float* p_Wo      = (const float*)d_in[14];
  const float* p_bo      = (const float*)d_in[15];
  const float* p_ln2_g   = (const float*)d_in[16];
  const float* p_ln2_b   = (const float*)d_in[17];
  const float* p_w1      = (const float*)d_in[18];
  const float* p_b1      = (const float*)d_in[19];
  const float* p_w2      = (const float*)d_in[20];
  const float* p_b2      = (const float*)d_in[21];
  const float* projW     = (const float*)d_in[22];
  const float* ca_Wq     = (const float*)d_in[23];
  const float* ca_Wk     = (const float*)d_in[24];
  const float* ca_Wv     = (const float*)d_in[25];
  const float* ca_Wo     = (const float*)d_in[26];
  const float* ca_bo     = (const float*)d_in[27];
  const float* cross_g   = (const float*)d_in[28];
  const float* cross_b   = (const float*)d_in[29];
  const float* ffd_w1    = (const float*)d_in[30];
  const float* ffd_b1    = (const float*)d_in[31];
  const float* ffd_w2    = (const float*)d_in[32];
  const float* ffd_b2    = (const float*)d_in[33];
  const float* out_g     = (const float*)d_in[34];
  const float* out_b     = (const float*)d_in[35];
  const float* head_W    = (const float*)d_in[36];
  const float* head_b    = (const float*)d_in[37];
  float* out = (float*)d_out;

  // -------- workspace layout --------
  float* w = (float*)d_ws;
  auto take = [&](long n) { float* p = w; w += n; return p; };
  float* x_enc  = take((long)B * SEQ * D);
  float* ybuf   = take((long)B * SEQ * D);
  float* qbuf   = take((long)B * SEQ * D);
  float* kbuf   = take((long)B * SEQ * D);
  float* vbuf   = take((long)B * SEQ * D);
  float* attnb  = take((long)B * SEQ * D);
  float* qp     = take((long)B * H * SEQ * MFP);
  float* kp     = take((long)B * H * SEQ * MFP);
  float* ksum   = take((long)B * H * MF);
  float* dinv   = take((long)B * H * SEQ);
  float* kv     = take((long)B * H * MFP * DHD);     // rows >= MF are never used
  float* ffbuf  = take((long)B * SEQ * FFD);
  float* dec_x  = take((long)B * SDQ * D);
  float* dec_y  = take((long)B * SDQ * D);
  float* qdec   = take((long)B * SDQ * D);
  float* ctx_o  = take((long)B * SDQ * D);
  float* scores = take((long)B * H * SDQ * SEQ);
  unsigned* gmax = (unsigned*)take(1);

  const int  ME = B * SEQ;        // 16384 encoder rows
  const int  MD = B * SDQ;        // 8192 decoder rows
  const int  BH = B * H;
  const float dn = 0.35355339059327373f;   // DH^-0.25

  // -------- embeddings --------
  launch_gemm(stream, 0, 0, 0, gp(in_enc, var_W, var_b, nullptr, x_enc, ME, D, IND, IND, D, D), 1);
  sinus_add_ln_kernel<<<ME, 256, 0, stream>>>(x_enc, expr, enc_g, enc_b);
  launch_gemm(stream, 0, 0, 0, gp(in_dec, var_W, var_b, nullptr, dec_x, MD, D, IND, IND, D, D), 1);

  // -------- Performer encoder layers --------
  for (int i = 0; i < DEPTH; ++i) {
    const float* Wq = p_Wq + (long)i * D * (H * DHD);
    const float* Wk = p_Wk + (long)i * D * (H * DHD);
    const float* Wv = p_Wv + (long)i * D * (H * DHD);
    const float* Wo = p_Wo + (long)i * (H * DHD) * D;
    const float* bo = p_bo + (long)i * D;
    const float* w1 = p_w1 + (long)i * D * FFD;
    const float* b1 = p_b1 + (long)i * FFD;
    const float* w2 = p_w2 + (long)i * FFD * D;
    const float* b2 = p_b2 + (long)i * D;
    const float* pr = projW + (long)i * MF * DHD;

    layernorm_kernel<<<ME, 256, 0, stream>>>(x_enc, nullptr, p_ln1_g + (long)i * D, p_ln1_b + (long)i * D, ybuf);
    launch_gemm(stream, 0, 0, 0, gp(ybuf, Wq, nullptr, nullptr, qbuf, ME, D, D, D, D, D), 1);
    launch_gemm(stream, 0, 0, 0, gp(ybuf, Wk, nullptr, nullptr, kbuf, ME, D, D, D, D, D), 1);
    launch_gemm(stream, 0, 0, 0, gp(ybuf, Wv, nullptr, nullptr, vbuf, ME, D, D, D, D, D), 1);

    // qd/kd: per-(b,h) [SEQ,DH] @ proj^T -> [SEQ,MF] (ldc = MFP), alpha = dn
    launch_gemm(stream, 0, 1, 0,
        gp(qbuf, pr, nullptr, nullptr, qp, SEQ, MF, DHD, D, DHD, MFP, dn, 0.f, H,
           (long)SEQ * D, DHD, 0, 0, (long)H * SEQ * MFP, (long)SEQ * MFP), B);
    launch_gemm(stream, 0, 1, 0,
        gp(kbuf, pr, nullptr, nullptr, kp, SEQ, MF, DHD, D, DHD, MFP, dn, 0.f, H,
           (long)SEQ * D, DHD, 0, 0, (long)H * SEQ * MFP, (long)SEQ * MFP), B);

    favor_q_kernel<<<BH * SEQ, 256, 0, stream>>>(qbuf, qp);
    gmax_init<<<1, 1, 0, stream>>>(gmax);
    kmax_kernel<<<4096, 256, 0, stream>>>(kp, BH * SEQ * MF, gmax);
    favor_k_kernel<<<BH * SEQ, 256, 0, stream>>>(kbuf, kp, gmax);
    ksum_kernel<<<BH, 512, 0, stream>>>(kp, ksum);
    dinv_kernel<<<BH * SEQ, 256, 0, stream>>>(qp, ksum, dinv);

    // kv[m,d] = sum_s kp[s,m] * v[s,d]   (A transposed, K = SEQ)
    launch_gemm(stream, 1, 0, 0,
        gp(kp, vbuf, nullptr, nullptr, kv, MF, DHD, SEQ, MFP, D, DHD, 1.f, 0.f, H,
           (long)H * SEQ * MFP, (long)SEQ * MFP, (long)SEQ * D, DHD,
           (long)H * MFP * DHD, (long)MFP * DHD), B);
    // attn[s,d] = (qp @ kv)[s,d] * d_inv[s]   (K = MFP; pad cols of qp are zero)
    launch_gemm(stream, 0, 0, 0,
        gp(qp, kv, nullptr, dinv, attnb, SEQ, DHD, MFP, MFP, DHD, D, 1.f, 0.f, H,
           (long)H * SEQ * MFP, (long)SEQ * MFP, (long)H * MFP * DHD, (long)MFP * DHD,
           (long)SEQ * D, DHD, (long)H * SEQ, SEQ), B);

    // x += attn @ Wo + bo
    launch_gemm(stream, 0, 0, 0, gp(attnb, Wo, bo, nullptr, x_enc, ME, D, D, D, D, D, 1.f, 1.f), 1);
    // FF block: x += gelu(LN(x) @ w1 + b1) @ w2 + b2
    layernorm_kernel<<<ME, 256, 0, stream>>>(x_enc, nullptr, p_ln2_g + (long)i * D, p_ln2_b + (long)i * D, ybuf);
    launch_gemm(stream, 0, 0, 1, gp(ybuf, w1, b1, nullptr, ffbuf, ME, FFD, D, D, FFD, FFD), 1);
    launch_gemm(stream, 0, 0, 0, gp(ffbuf, w2, b2, nullptr, x_enc, ME, D, FFD, FFD, D, D, 1.f, 1.f), 1);
  }

  // -------- decoder: softmax cross-attention over encoder context --------
  layernorm_kernel<<<MD, 256, 0, stream>>>(dec_x, nullptr, dec_g, dec_b, dec_y);
  launch_gemm(stream, 0, 0, 0, gp(dec_y, ca_Wq, nullptr, nullptr, qdec, MD, D, D, D, D, D), 1);
  launch_gemm(stream, 0, 0, 0, gp(x_enc, ca_Wk, nullptr, nullptr, kbuf, ME, D, D, D, D, D), 1);
  launch_gemm(stream, 0, 0, 0, gp(x_enc, ca_Wv, nullptr, nullptr, vbuf, ME, D, D, D, D, D), 1);

  // scores[q,k] = (Q @ K^T) * DH^-0.5 per (b,h)
  launch_gemm(stream, 0, 1, 0,
      gp(qdec, kbuf, nullptr, nullptr, scores, SDQ, SEQ, DHD, D, D, SEQ, 0.125f, 0.f, H,
         (long)SDQ * D, DHD, (long)SEQ * D, DHD, (long)H * SDQ * SEQ, (long)SDQ * SEQ), B);
  softmax_kernel<<<BH * SDQ, 256, 0, stream>>>(scores);
  // o_h = softmax(scores) @ V  -> merged into dec_x (reuse)
  launch_gemm(stream, 0, 0, 0,
      gp(scores, vbuf, nullptr, nullptr, dec_x, SDQ, DHD, SEQ, SEQ, D, D, 1.f, 0.f, H,
         (long)H * SDQ * SEQ, (long)SDQ * SEQ, (long)SEQ * D, DHD, (long)SDQ * D, DHD), B);
  // o = merged @ Wo + bo
  launch_gemm(stream, 0, 0, 0, gp(dec_x, ca_Wo, ca_bo, nullptr, ctx_o, MD, D, D, D, D, D), 1);
  // out = LN(o + dec)
  layernorm_kernel<<<MD, 256, 0, stream>>>(ctx_o, dec_y, cross_g, cross_b, qdec);
  // out = gelu(out @ w1 + b1) @ w2 + b2  (no residual)
  launch_gemm(stream, 0, 0, 1, gp(qdec, ffd_w1, ffd_b1, nullptr, ffbuf, MD, FFD, D, D, FFD, FFD), 1);
  launch_gemm(stream, 0, 0, 0, gp(ffbuf, ffd_w2, ffd_b2, nullptr, ctx_o, MD, D, FFD, FFD, D, D), 1);
  // out = LN(out)
  layernorm_kernel<<<MD, 256, 0, stream>>>(ctx_o, nullptr, out_g, out_b, dec_x);
  // logits = out @ head_W + head_b
  launch_gemm(stream, 0, 0, 0, gp(dec_x, head_W, head_b, nullptr, out, MD, NB, D, D, NB, NB), 1);
}
